// TFN_grid_density_57698590655145
// MI455X (gfx1250) — compile-verified, matrix operand order checked by reading, not run on hardware
//
#include <hip/hip_runtime.h>
#include <hip/hip_fp16.h>
#include <math.h>

typedef __attribute__((ext_vector_type(16))) _Float16 v16h;
typedef __attribute__((ext_vector_type(8)))  float    v8f;
typedef unsigned int u32x4 __attribute__((ext_vector_type(4)));
typedef int          i32x4 __attribute__((ext_vector_type(4)));
typedef int          i32x8 __attribute__((ext_vector_type(8)));

#define K_NEIGH 32
#define GS_CONST 6.23832462504f   /* ln2 * SHELLS^2, SHELLS=3 */
#define CDIV(a,b) (((a)+(b)-1)/(b))

// ---------------------------------------------------------------- fragment-order addressing
// ISA VGPR layouts (CDNA5 ISA): lanes 0-15 hold K {0..7,16..23}, lanes 16-31 hold
// K {8..15,24..31}. We store operands pre-swizzled so lane's 16 halves are contiguous (32B):
//   A-frag:  off = ((m>>4)*KT + (k>>5))*512 + lane*16 + j
//   B-frag:  off = ((k>>5)*NT + (n>>4))*512 + lane*16 + j
__device__ __forceinline__ long afrag_off(int m, int k, int KT) {
  int kr = k & 31;
  int j    = (kr & 7) + ((kr & 16) ? 8 : 0);
  int lane = (m & 15) + (((kr >> 3) & 1) << 4);
  return ((long)((m >> 4)*KT + (k >> 5))*32 + lane)*16 + j;
}
__device__ __forceinline__ long bfrag_off(int k, int n, int NT) {
  int kr = k & 31;
  int j    = (kr & 7) + ((kr & 16) ? 8 : 0);
  int lane = (n & 15) + (((kr >> 3) & 1) << 4);
  return ((long)((k >> 5)*NT + (n >> 4))*32 + lane)*16 + j;
}

// ---------------------------------------------------------------- SH basis
__device__ __forceinline__ void sh16(float x, float y, float z, float* Y) {
  float x2=x*x, y2=y*y, z2=z*z;
  Y[0]=0.28209479177387814f;
  Y[1]=0.4886025119029199f*y;
  Y[2]=0.4886025119029199f*z;
  Y[3]=0.4886025119029199f*x;
  Y[4]=1.0925484305920792f*x*y;
  Y[5]=1.0925484305920792f*y*z;
  Y[6]=0.31539156525252005f*(3.0f*z2-1.0f);
  Y[7]=1.0925484305920792f*x*z;
  Y[8]=0.5462742152960396f*(x2-y2);
  Y[9]=0.5900435899266435f*y*(3.0f*x2-y2);
  Y[10]=2.890611442640554f*x*y*z;
  Y[11]=0.4570457994644658f*y*(5.0f*z2-1.0f);
  Y[12]=0.3731763325901154f*z*(5.0f*z2-3.0f);
  Y[13]=0.4570457994644658f*x*(5.0f*z2-1.0f);
  Y[14]=1.445305721320277f*z*(x2-y2);
  Y[15]=0.5900435899266435f*x*(x2-3.0f*y2);
}

// YB f32 (64x16); YBhF: A-frag 64x32 (K 16..31 zero); YBTF: A-frag 16x64 (transpose)
__global__ void yb_kernel(float* __restrict__ YB, _Float16* __restrict__ YBhF,
                          _Float16* __restrict__ YBTF) {
  int i = threadIdx.x;
  if (i >= 64) return;
  float ga = 3.14159265358979323846f * (3.0f - sqrtf(5.0f));
  float z = 1.0f - (2.0f*(float)i + 1.0f)/64.0f;
  float r = sqrtf(fmaxf(0.0f, 1.0f - z*z));
  float th = ga * (float)i;
  float Y[16];
  sh16(r*__cosf(th), r*__sinf(th), z, Y);
  for (int q=0;q<16;++q) {
    YB[i*16+q] = Y[q];
    YBhF[afrag_off(i, q,      1)] = (_Float16)Y[q];
    YBhF[afrag_off(i, q + 16, 1)] = (_Float16)0.f;   // K pad
    YBTF[afrag_off(q, i,      2)] = (_Float16)Y[q];  // 16x64, KT=2
  }
}

// ---------------------------------------------------------------- pooling
__global__ void pool_kernel(const float* __restrict__ src, float* __restrict__ dst,
                            int B, int H, int C, int isMax) {
  long t = (long)blockIdx.x*blockDim.x + threadIdx.x;
  long total = (long)B*H*H*H*C;
  if (t >= total) return;
  int c = (int)(t % C); long u = t / C;
  int d = (int)(u % H); u /= H;
  int w = (int)(u % H); u /= H;
  int h = (int)(u % H); int b = (int)(u / H);
  int H2 = 2*H;
  float acc = isMax ? -3.4e38f : 0.f;
  for (int dh=0; dh<2; ++dh)
    for (int dw=0; dw<2; ++dw)
      for (int dd=0; dd<2; ++dd) {
        long s = ((((long)b*H2 + (2*h+dh))*H2 + (2*w+dw))*H2 + (2*d+dd))*C + c;
        float v = src[s];
        acc = isMax ? fmaxf(acc, v) : (acc + v);
      }
  dst[t] = isMax ? acc : acc * 0.125f;
}

// ---------------------------------------------------------------- gradient (jnp.gradient semantics)
__global__ void grad_kernel(const float* __restrict__ den, float* __restrict__ g,
                            int B, int H) {
  long t = (long)blockIdx.x*blockDim.x + threadIdx.x;
  long total = (long)B*H*H*H;
  if (t >= total) return;
  long u = t;
  int d = (int)(u % H); u /= H;
  int w = (int)(u % H); u /= H;
  int h = (int)(u % H); int b = (int)(u / H);
  const float* db = den + (long)b*H*H*H;
  auto val = [&](int hh,int ww,int dd) { return db[((long)hh*H+ww)*H+dd]; };
  float gh = (h==0)   ? val(1,w,d)-val(0,w,d)
           : (h==H-1) ? val(H-1,w,d)-val(H-2,w,d)
                      : 0.5f*(val(h+1,w,d)-val(h-1,w,d));
  float gw = (w==0)   ? val(h,1,d)-val(h,0,d)
           : (w==H-1) ? val(h,H-1,d)-val(h,H-2,d)
                      : 0.5f*(val(h,w+1,d)-val(h,w-1,d));
  float gd = (d==0)   ? val(h,w,1)-val(h,w,0)
           : (d==H-1) ? val(h,w,H-1)-val(h,w,H-2)
                      : 0.5f*(val(h,w,d+1)-val(h,w,d-1));
  g[t*3+0]=gh; g[t*3+1]=gw; g[t*3+2]=gd;
}

__global__ void feat0_kernel(const float* __restrict__ den, const float* __restrict__ g,
                             float* __restrict__ feat, long n) {
  long t = (long)blockIdx.x*blockDim.x + threadIdx.x;
  if (t >= n) return;
  feat[t*4+0]=den[t];
  feat[t*4+1]=g[t*3+0];
  feat[t*4+2]=g[t*3+1];
  feat[t*4+3]=g[t*3+2];
}

// ---------------------------------------------------------------- ball group + SH kernels
__global__ void ball_kernel(const float* __restrict__ src, const float* __restrict__ tgt,
                            int N, int M, float radius,
                            int* __restrict__ idxOut, float* __restrict__ kernOut) {
  __shared__ float rd[256]; __shared__ int ri[256];
  __shared__ float seld[K_NEIGH]; __shared__ int seli[K_NEIGH];
  int m = blockIdx.x, b = blockIdx.y, tid = threadIdx.x;
  const float* tp = tgt + ((long)b*M + m)*3;
  float tx=tp[0], ty=tp[1], tz=tp[2];
  const float* sp = src + (long)b*N*3;
  float lastd = -1.f; int lasti = -1;
  for (int kk=0; kk<K_NEIGH; ++kk) {
    float bd = 3.4e38f; int bi = 0x7fffffff;
    for (int j=tid; j<N; j+=256) {
      float dx=sp[j*3]-tx, dy=sp[j*3+1]-ty, dz=sp[j*3+2]-tz;
      float d2 = dx*dx+dy*dy+dz*dz;
      bool adm = (d2 > lastd) || (d2 == lastd && j > lasti);
      if (adm && (d2 < bd || (d2 == bd && j < bi))) { bd=d2; bi=j; }
    }
    rd[tid]=bd; ri[tid]=bi;
    __syncthreads();
    for (int s=128; s>0; s>>=1) {
      if (tid < s) {
        if (rd[tid+s] < rd[tid] || (rd[tid+s]==rd[tid] && ri[tid+s]<ri[tid]))
          { rd[tid]=rd[tid+s]; ri[tid]=ri[tid+s]; }
      }
      __syncthreads();
    }
    lastd = rd[0]; lasti = ri[0];
    if (tid==0) { seld[kk]=lastd; seli[kk]=lasti; }
    __syncthreads();
  }
  if (tid < K_NEIGH) {
    int j = seli[tid];
    long base = ((long)b*M + m)*K_NEIGH + tid;
    idxOut[base] = j;
    float dx=sp[j*3]-tx, dy=sp[j*3+1]-ty, dz=sp[j*3+2]-tz;
    float d2 = fmaxf(seld[tid], 1e-12f);
    float dist = sqrtf(d2);
    float mask = (dist < radius) ? 1.f : 0.f;
    float inv = 1.f/(dist + 1e-8f);
    float Y[16]; sh16(dx*inv, dy*inv, dz*inv, Y);
    float dn = dist/radius;
    float t0=dn, t1=dn-0.5f, t2=dn-1.0f;
    float rad[3];
    rad[0]=__expf(-GS_CONST*t0*t0)*mask;
    rad[1]=__expf(-GS_CONST*t1*t1)*mask;
    rad[2]=__expf(-GS_CONST*t2*t2)*mask;
    float* kp = kernOut + base*48;
    for (int p=0;p<16;++p)
      for (int s2=0;s2<3;++s2)
        kp[p*3+s2] = Y[p]*rad[s2];
  }
}

// ---------------------------------------------------------------- conv (neighbor contraction) + eq mix
// Emits qh as a per-point B-frag matrix (32 x U, rows 16..31 zero).
__global__ void conv_mix_kernel(
    const float* __restrict__ feat, int fstride,
    const int* __restrict__ idx, const float* __restrict__ kern,
    const float* __restrict__ gradT, const float* __restrict__ denT,
    const float* __restrict__ eqw0, const float* __restrict__ eqw1,
    const float* __restrict__ eqw2, const float* __restrict__ eqw3,
    int c0, int c1, int c2, int c3,
    int e0, int e1, int e2, int e3,
    int N, int M, int U,
    _Float16* __restrict__ qh)
{
  __shared__ float pre[6144];
  __shared__ int nb[K_NEIGH];
  int m = blockIdx.x, b = blockIdx.y, tid = threadIdx.x;
  long pt = (long)b*M + m;
  const float* kp = kern + pt*K_NEIGH*48;
  if (tid < K_NEIGH) nb[tid] = idx[pt*K_NEIGH + tid];
  __syncthreads();
  int cl[4]   = {c0,c1,c2,c3};
  int el[4]   = {e0,e1,e2,e3};
  int offl[4] = {0, c0, c0+3*c1, c0+3*c1+5*c2};
  int bl[4];  bl[0]=0; bl[1]=e0; bl[2]=e0+3*e1; bl[3]=e0+3*e1+5*e2;
  int aoff[4] = {0,1,4,9};
  const float invK = 1.0f/(float)K_NEIGH;
  const float* fb = feat + (long)b*N*fstride;
  for (int l=0; l<4; ++l) {
    int rows = 2*l+1;
    int cA = 3*c0;
    int cB = (l>0) ? 3*cl[l] : 0;
    int tot = rows*(cA+cB);
    for (int t=tid; t<tot; t+=blockDim.x) {
      int p = t/(cA+cB), cc = t%(cA+cB);
      float acc = 0.f;
      if (cc < cA) {
        int c = cc/3, s = cc%3;
        for (int k=0;k<K_NEIGH;++k)
          acc += fb[(long)nb[k]*fstride + c] * kp[k*48 + (aoff[l]+p)*3 + s];
      } else {
        int c = (cc-cA)/3, s = (cc-cA)%3;
        for (int k=0;k<K_NEIGH;++k)
          acc += fb[(long)nb[k]*fstride + offl[l] + p*cl[l] + c] * kp[k*48 + s];
      }
      pre[bl[l] + p*el[l] + cc] = acc*invK;
    }
  }
  if (tid < 3) pre[bl[1] + tid*e1 + (e1-1)] = gradT[pt*3 + tid]; // grad append to l=1
  __syncthreads();
  float w = denT[pt];
  const float* eqw[4] = {eqw0, eqw1, eqw2, eqw3};
  _Float16* qp = qh + pt*32*(long)U;
  int NTq = U >> 4;
  for (int t=tid; t<16*U; t+=blockDim.x) {
    int pg = t/U, u = t%U;
    int l = (pg>=9)?3:(pg>=4)?2:(pg>=1)?1:0;
    int p = pg - aoff[l];
    const float* pr = pre + bl[l] + p*el[l];
    const float* wq = eqw[l];
    float acc = 0.f;
    for (int c=0;c<el[l];++c) acc += pr[c]*wq[c*U+u];
    qp[bfrag_off(pg, u, NTq)] = (_Float16)(acc*w);
  }
  for (int t=tid; t<16*U; t+=blockDim.x) {
    int k = 16 + t/U, u = t%U;
    qp[bfrag_off(k, u, NTq)] = (_Float16)0.f;     // K pad rows
  }
}

// ---------------------------------------------------------------- Tensor Data Mover: linear f16 -> LDS
__device__ __forceinline__ void tdm_load_to_lds(unsigned ldsoff, const void* gaddr,
                                                int ncols, int nrows) {
  unsigned long long ga = (unsigned long long)gaddr;
  u32x4 g0;
  g0[0] = 1u;                                             // count=1 (valid user D#)
  g0[1] = ldsoff;                                         // lds_addr (bytes)
  g0[2] = (unsigned)(ga & 0xffffffffu);                   // global_addr[31:0]
  g0[3] = (unsigned)((ga >> 32) & 0x01ffffffu)            // global_addr[56:32]
        | 0x80000000u;                                    // type = 2 ("image")
  unsigned long long q0 = (1ull << 16)                                      // data_size = 2 bytes
                        | ((unsigned long long)((unsigned)ncols & 0xffffu) << 48); // tensor_dim0 lo16
  unsigned long long q1 = (((unsigned long long)(unsigned)ncols >> 16) & 0xffffull)   // tensor_dim0 hi16
                        | ((unsigned long long)(unsigned)nrows << 16)                 // tensor_dim1
                        | ((unsigned long long)((unsigned)ncols & 0xffffu) << 48);    // tile_dim0
  unsigned long long q2 = ((unsigned long long)((unsigned)nrows & 0xffffu))           // tile_dim1
                        | ((unsigned long long)(unsigned)ncols << 32);                // dim0_stride lo32
  unsigned long long q3 = 0ull;
  i32x8 g1;
  g1[0]=(int)(unsigned)(q0 & 0xffffffffull); g1[1]=(int)(unsigned)(q0 >> 32);
  g1[2]=(int)(unsigned)(q1 & 0xffffffffull); g1[3]=(int)(unsigned)(q1 >> 32);
  g1[4]=(int)(unsigned)(q2 & 0xffffffffull); g1[5]=(int)(unsigned)(q2 >> 32);
  g1[6]=(int)(unsigned)(q3 & 0xffffffffull); g1[7]=(int)(unsigned)(q3 >> 32);
  i32x4 z4 = {0,0,0,0};
#if __clang_major__ >= 23
  i32x8 z8 = {0,0,0,0,0,0,0,0};
  __builtin_amdgcn_tensor_load_to_lds(g0, g1, z4, z4, z8, 0);
#else
  __builtin_amdgcn_tensor_load_to_lds(g0, g1, z4, z4, 0);
#endif
}

// ---------------------------------------------------------------- frag-direct GEMM (1 wave / tile)
// A, B pre-swizzled in fragment order in global; each fragment = one contiguous v16h per lane.
__global__ void __launch_bounds__(32) frag_gemm_kernel(
    const _Float16* __restrict__ A, long sA,
    const _Float16* __restrict__ B, long sB,
    float* __restrict__ D, long sD,
    int NT, int KT, float alpha)
{
  int tn = blockIdx.x, tm = blockIdx.y, batch = blockIdx.z;
  int lane = threadIdx.x;
  const _Float16* Ab = A + (long)batch*sA;
  const _Float16* Bb = B + (long)batch*sB;
  if (KT > 1) __builtin_prefetch(Bb + ((long)(NT + tn)*32 + lane)*16, 0, 1);
  v8f acc = {};
  for (int kt=0; kt<KT; ++kt) {
    v16h a = *(const v16h*)(Ab + ((long)(tm*KT + kt)*32 + lane)*16);
    v16h b = *(const v16h*)(Bb + ((long)(kt*NT + tn)*32 + lane)*16);
    acc = __builtin_amdgcn_wmma_f32_16x16x32_f16(false, a, false, b,
                                                 (short)0, acc, false, false);
  }
  int n = lane & 15, mb = (lane >> 4) << 3;
  int N = NT*16;
#pragma unroll
  for (int r=0;r<8;++r)
    D[(long)batch*sD + (long)(tm*16 + mb + r)*N + tn*16 + n] = acc[r]*alpha;
}

// ---------------------------------------------------------------- MLP GEMM: 4 waves / point, TDM weights
// Weights (frag-order, Kdim*N f16) are DMA'd to LDS by wave 0 via TENSOR_LOAD_TO_LDS; all LDS
// fragment reads are contiguous v16h (2 x ds_load_b128). Activations are frag-direct from global.
// dh_mode: 0=no f16 out, 1=write A-frag (operand for next GEMM over N), 2=write B-frag (K=row).
__global__ void __launch_bounds__(128) mlp_gemm_kernel(
    const _Float16* __restrict__ A, long sA,
    const _Float16* __restrict__ W,
    float* __restrict__ D, long sD,
    _Float16* __restrict__ Dh, long sDh, int dh_mode,
    const float* __restrict__ bias, int N, int Kdim, int relu)
{
  extern __shared__ _Float16 Ws[];
  int wave = threadIdx.x >> 5, lane = threadIdx.x & 31;
  int batch = blockIdx.x;
  int NT = N >> 4, KT = Kdim >> 5;
  if (wave == 0) {
    tdm_load_to_lds((unsigned)(unsigned long long)(void*)Ws, (const void*)W, Kdim*N, 1);
    __builtin_amdgcn_s_wait_tensorcnt(0);
  }
  const _Float16* Ab = A + (long)batch*sA + (long)wave*KT*512;
  __builtin_prefetch(Ab + lane*16, 0, 1);
  __syncthreads();

  for (int tn = 0; tn < NT; ++tn) {
    v8f acc = {};
    for (int kt = 0; kt < KT; kt += 2) {
      v16h a0 = *(const v16h*)(Ab + ((long)kt*32 + lane)*16);
      v16h b0 = *(const v16h*)(Ws + ((long)(kt*NT + tn)*32 + lane)*16);
      acc = __builtin_amdgcn_wmma_f32_16x16x32_f16(false, a0, false, b0,
                                                   (short)0, acc, false, false);
      if (kt + 1 < KT) {
        v16h a1 = *(const v16h*)(Ab + ((long)(kt+1)*32 + lane)*16);
        v16h b1 = *(const v16h*)(Ws + ((long)((kt+1)*NT + tn)*32 + lane)*16);
        acc = __builtin_amdgcn_wmma_f32_16x16x32_f16(false, a1, false, b1,
                                                     (short)0, acc, false, false);
      }
    }
    int n = lane & 15, mb = (lane >> 4) << 3;
#pragma unroll
    for (int r=0;r<8;++r) {
      float v = acc[r];
      int col = tn*16 + n;
      int row = wave*16 + mb + r;
      if (bias) v += bias[col];
      if (relu) v = fmaxf(v, 0.f);
      if (D) D[(long)batch*sD + (long)row*N + col] = v;
      if (dh_mode == 1)      Dh[(long)batch*sDh + afrag_off(row, col, N >> 5)] = (_Float16)v;
      else if (dh_mode == 2) Dh[(long)batch*sDh + bfrag_off(row, col, N >> 4)] = (_Float16)v;
    }
  }
}

// ---------------------------------------------------------------- misc elementwise
// weights f32 (Kdim x N row-major) -> f16 B-frag order
__global__ void cvt_wswz_kernel(const float* __restrict__ s, _Float16* __restrict__ d,
                                int Kdim, int N) {
  long t = (long)blockIdx.x*blockDim.x + threadIdx.x;
  long total = (long)Kdim*N;
  if (t >= total) return;
  int k = (int)(t / N), n = (int)(t % N);
  d[bfrag_off(k, n, N >> 4)] = (_Float16)s[t];
}

__global__ void stats_kernel(const float* __restrict__ f, long rows, int U,
                             float* __restrict__ mu, float* __restrict__ var) {
  __shared__ float s1[256], s2[256];
  int u = blockIdx.x, tid = threadIdx.x;
  float a=0.f, b=0.f;
  for (long r=tid; r<rows; r+=256) { float v=f[r*(long)U+u]; a+=v; b+=v*v; }
  s1[tid]=a; s2[tid]=b;
  __syncthreads();
  for (int s=128; s>0; s>>=1) {
    if (tid<s) { s1[tid]+=s1[tid+s]; s2[tid]+=s2[tid+s]; }
    __syncthreads();
  }
  if (tid==0) { float m=s1[0]/(float)rows; mu[u]=m; var[u]=s2[0]/(float)rows - m*m; }
}

// BN + ReLU; emits f16 in per-point A-frag order (64 x U)
__global__ void bn_relu_frag_kernel(const float* __restrict__ f,
                                    const float* __restrict__ mu, const float* __restrict__ var,
                                    const float* __restrict__ g, const float* __restrict__ bb,
                                    _Float16* __restrict__ fh, long n, int U) {
  long t = (long)blockIdx.x*blockDim.x + threadIdx.x;
  if (t >= n) return;
  int u = (int)(t % U);
  long rw = t / U;
  int row = (int)(rw % 64);
  long ptI = rw / 64;
  float v = (f[t]-mu[u])*rsqrtf(var[u]+1e-5f)*g[u]+bb[u];
  fh[ptI*64*(long)U + afrag_off(row, u, U >> 5)] = (_Float16)fmaxf(v, 0.f);
}

__global__ void scatter_feat_kernel(const float* __restrict__ c, const float* __restrict__ grad,
                                    float* __restrict__ feat, long npts, int U, int fstride) {
  long t = (long)blockIdx.x*blockDim.x + threadIdx.x;
  long total = npts*(long)fstride;
  if (t >= total) return;
  long p = t / fstride; int pos = (int)(t % fstride);
  const float* cp = c + p*16*(long)U;
  float v;
  int b1 = U, b2 = U + 3*(U+1), b3 = U + 3*(U+1) + 5*U;
  if (pos < b1) v = cp[pos];
  else if (pos < b2) { int q=pos-b1; int pr=q/(U+1), cc=q%(U+1);
                       v = (cc<U) ? cp[(1+pr)*U+cc] : grad[p*3+pr]; }
  else if (pos < b3) { int q=pos-b2; int pr=q/U, cc=q%U; v = cp[(4+pr)*U+cc]; }
  else               { int q=pos-b3; int pr=q/U, cc=q%U; v = cp[(9+pr)*U+cc]; }
  feat[t] = v;
}

__global__ void max_kernel(const float* __restrict__ h2, float* __restrict__ out,
                           int B, int M, int SU) {
  long t = (long)blockIdx.x*blockDim.x + threadIdx.x;
  long total = (long)B*SU;
  if (t >= total) return;
  int b = (int)(t / SU), j = (int)(t % SU);
  float acc = -3.4e38f;
  for (int m=0; m<M; ++m)
    acc = fmaxf(acc, h2[((long)b*M + m)*(long)SU + j]);
  out[t] = acc;
}

// ---------------------------------------------------------------- host orchestration
extern "C" void kernel_launch(void* const* d_in, const int* in_sizes, int n_in,
                              void* d_out, int out_size, void* d_ws, size_t ws_size,
                              hipStream_t stream) {
  (void)in_sizes; (void)n_in; (void)out_size; (void)ws_size;
  const float* x_grid = (const float*)d_in[0];
  const float* x_dens = (const float*)d_in[1];
  const float* eqw[3][4];
  for (int i=0;i<3;++i) for (int l=0;l<4;++l) eqw[i][l] = (const float*)d_in[2+4*i+l];
  const float *W1[3], *b1[3], *W2[3], *b2[3];
  for (int i=0;i<3;++i) {
    W1[i]=(const float*)d_in[14+4*i+0]; b1[i]=(const float*)d_in[14+4*i+1];
    W2[i]=(const float*)d_in[14+4*i+2]; b2[i]=(const float*)d_in[14+4*i+3];
  }
  const float *bng[3], *bnb[3];
  for (int i=0;i<3;++i) { bng[i]=(const float*)d_in[26+i]; bnb[i]=(const float*)d_in[29+i]; }

  const int B = 4;
  const int nlev[4] = {13824, 1728, 216, 27};
  const int Hlev[4] = {24, 12, 6, 3};
  const float radii[3] = {0.2f, 0.4f, 0.8f};
  const int U0a[3] = {32, 64, 128};
  const int U1a[3] = {32, 64, 128};
  const int U2a[3] = {32, 64, 256};
  const int eqin[3][4] = {{3,7,3,3},{96,196,192,192},{192,388,384,384}};
  const int fstr[3] = {4, 16*32+3, 16*64+3};

  char* cur = (char*)d_ws;
  auto alloc = [&](size_t bytes)->void* {
    void* p = (void*)cur; cur += (bytes + 255) & ~(size_t)255; return p;
  };

  float*    YB   = (float*)   alloc(64*16*sizeof(float));
  _Float16* YBhF = (_Float16*)alloc(64*32*sizeof(_Float16));
  _Float16* YBTF = (_Float16*)alloc(16*64*sizeof(_Float16));
  float* pts[4]; pts[0] = (float*)x_grid;
  float* den[4]; den[0] = (float*)x_dens;
  for (int l=1;l<4;++l) {
    pts[l] = (float*)alloc((size_t)B*nlev[l]*3*sizeof(float));
    den[l] = (float*)alloc((size_t)B*nlev[l]*sizeof(float));
  }
  float* grd[4];
  for (int l=0;l<4;++l) grd[l] = (float*)alloc((size_t)B*nlev[l]*3*sizeof(float));
  float* feat[3];
  feat[0] = (float*)alloc((size_t)B*nlev[0]*fstr[0]*sizeof(float));
  feat[1] = (float*)alloc((size_t)B*nlev[1]*fstr[1]*sizeof(float));
  feat[2] = (float*)alloc((size_t)B*nlev[2]*fstr[2]*sizeof(float));

  const size_t PTSmax = (size_t)B*1728;
  int*      idxB  = (int*)     alloc(PTSmax*K_NEIGH*sizeof(int));
  float*    kernB = (float*)   alloc(PTSmax*K_NEIGH*48*sizeof(float));
  _Float16* qhB   = (_Float16*)alloc(PTSmax*32*32*sizeof(_Float16));
  float*    fB    = (float*)   alloc(PTSmax*64*32*sizeof(float));
  _Float16* fhB   = (_Float16*)alloc(PTSmax*64*32*sizeof(_Float16));
  _Float16* h1hB  = (_Float16*)alloc(PTSmax*64*32*sizeof(_Float16));
  float*    h2B   = (float*)   alloc(PTSmax*64*32*sizeof(float));
  _Float16* h2hB  = (_Float16*)alloc(PTSmax*64*32*sizeof(_Float16));
  float*    cB    = (float*)   alloc(PTSmax*16*32*sizeof(float));
  float*    muB   = (float*)   alloc(256*sizeof(float));
  float*    varB  = (float*)   alloc(256*sizeof(float));
  _Float16* W1h   = (_Float16*)alloc(128*256*sizeof(_Float16));
  _Float16* W2h   = (_Float16*)alloc(128*256*sizeof(_Float16));

  yb_kernel<<<1, 64, 0, stream>>>(YB, YBhF, YBTF);

  for (int l=1; l<4; ++l) {
    int H = Hlev[l];
    long t3 = (long)B*H*H*H*3;
    pool_kernel<<<(unsigned)CDIV(t3,256), 256, 0, stream>>>(pts[l-1], pts[l], B, H, 3, 0);
    long t1 = (long)B*H*H*H;
    pool_kernel<<<(unsigned)CDIV(t1,256), 256, 0, stream>>>(den[l-1], den[l], B, H, 1, 1);
  }
  for (int l=0; l<4; ++l) {
    long t = (long)B*nlev[l];
    grad_kernel<<<(unsigned)CDIV(t,256), 256, 0, stream>>>(den[l], grd[l], B, Hlev[l]);
  }
  {
    long t = (long)B*nlev[0];
    feat0_kernel<<<(unsigned)CDIV(t,256), 256, 0, stream>>>(den[0], grd[0], feat[0], t);
  }

  for (int i=0; i<3; ++i) {
    int N = nlev[i], M = nlev[i+1];
    int U0 = U0a[i], U1 = U1a[i], U2 = U2a[i];
    long PTS = (long)B*M;

    ball_kernel<<<dim3((unsigned)M,(unsigned)B), 256, 0, stream>>>(
        pts[i], pts[i+1], N, M, radii[i], idxB, kernB);

    int c0,c1,c2,c3;
    if (i==0) { c0=1; c1=1; c2=0; c3=0; }
    else      { int Up=U2a[i-1]; c0=Up; c1=Up+1; c2=Up; c3=Up; }
    conv_mix_kernel<<<dim3((unsigned)M,(unsigned)B), 128, 0, stream>>>(
        feat[i], fstr[i], idxB, kernB, grd[i+1], den[i+1],
        eqw[i][0], eqw[i][1], eqw[i][2], eqw[i][3],
        c0,c1,c2,c3, eqin[i][0], eqin[i][1], eqin[i][2], eqin[i][3],
        N, M, U0, qhB);

    // f = YB x coeffs  (64 x U0, K=32 incl. pad); frag-direct operands
    frag_gemm_kernel<<<dim3((unsigned)(U0/16), 4, (unsigned)PTS), 32, 0, stream>>>(
        YBhF, 0, qhB, (long)32*U0, fB, (long)64*U0, U0/16, 1, 1.0f);

    stats_kernel<<<(unsigned)U0, 256, 0, stream>>>(fB, PTS*64, U0, muB, varB);
    {
      long n = PTS*64*(long)U0;
      bn_relu_frag_kernel<<<(unsigned)CDIV(n,256), 256, 0, stream>>>(
          fB, muB, varB, bng[i], bnb[i], fhB, n, U0);
    }

    cvt_wswz_kernel<<<(unsigned)CDIV((long)U0*U1,256), 256, 0, stream>>>(W1[i], W1h, U0, U1);
    cvt_wswz_kernel<<<(unsigned)CDIV((long)U1*U2,256), 256, 0, stream>>>(W2[i], W2h, U1, U2);

    // h1 = relu(f x W1 + b1): TDM weights in LDS; h1 written as A-frag for next GEMM
    mlp_gemm_kernel<<<dim3((unsigned)PTS), 128, (size_t)U0*U1*sizeof(_Float16), stream>>>(
        fhB, (long)64*U0, W1h, nullptr, 0, h1hB, (long)64*U1, 1, b1[i], U1, U0, 1);

    // h2 = relu(h1 x W2 + b2): f32 for max/stats path, f16 as B-frag for back-projection
    mlp_gemm_kernel<<<dim3((unsigned)PTS), 128, (size_t)U1*U2*sizeof(_Float16), stream>>>(
        h1hB, (long)64*U1, W2h, h2B, (long)64*U2, h2hB, (long)64*U2, 2, b2[i], U2, U1, 1);

    if (i < 2) {
      // c = (4pi/S) * YB^T x h2   (16 x U2, K=64); frag-direct operands
      frag_gemm_kernel<<<dim3((unsigned)(U2/16), 1, (unsigned)PTS), 32, 0, stream>>>(
          YBTF, 0, h2hB, (long)64*U2, cB, (long)16*U2, U2/16, 2,
          4.0f*3.14159265358979f/64.0f);
      long tot = PTS*(long)fstr[i+1];
      scatter_feat_kernel<<<(unsigned)CDIV(tot,256), 256, 0, stream>>>(
          cB, grd[i+1], feat[i+1], PTS, U2, fstr[i+1]);
    } else {
      long tot = (long)B*64*U2;
      max_kernel<<<(unsigned)CDIV(tot,256), 256, 0, stream>>>(
          h2B, (float*)d_out, B, M, 64*U2);
    }
  }
}
